// Sampler_24378234372773
// MI455X (gfx1250) — compile-verified, matrix operand order checked by reading, not run on hardware
//
#include <hip/hip_runtime.h>
#include <hip/hip_bf16.h>

#define B_ 256
#define V_ 32000
#define H_ 4096
#define L_ 128

typedef __attribute__((ext_vector_type(16))) __bf16 v16bf;
typedef __attribute__((ext_vector_type(8)))  float  v8f;
typedef __attribute__((ext_vector_type(4)))  unsigned u32x4;
typedef __attribute__((ext_vector_type(8)))  unsigned u32x8;

// Native f32 -> bf16 (lowers to v_cvt_pk_bf16_f32 on gfx1250)
__device__ inline __bf16 f2bf(float f) { return (__bf16)f; }

// Build one 16x32 bf16 WMMA fragment for this lane from a K-contiguous fp32 row.
// Per CDNA5 ISA layout: lanes 0-15 = M/N row, half = lane>>4 selects K sub-halves;
// elements 0..7  <- K = half*8 + 0..7      (contiguous, 2x b128)
// elements 8..15 <- K = 16 + half*8 + 0..7 (contiguous, 2x b128)
__device__ inline v16bf make_frag(const float* __restrict__ p, int half) {
    const float4* q0 = reinterpret_cast<const float4*>(p + half * 8);
    const float4* q1 = reinterpret_cast<const float4*>(p + 16 + half * 8);
    float4 x0 = q0[0], x1 = q0[1], y0 = q1[0], y1 = q1[1];
    v16bf f;
    f[0]  = f2bf(x0.x); f[1]  = f2bf(x0.y); f[2]  = f2bf(x0.z); f[3]  = f2bf(x0.w);
    f[4]  = f2bf(x1.x); f[5]  = f2bf(x1.y); f[6]  = f2bf(x1.z); f[7]  = f2bf(x1.w);
    f[8]  = f2bf(y0.x); f[9]  = f2bf(y0.y); f[10] = f2bf(y0.z); f[11] = f2bf(y0.w);
    f[12] = f2bf(y1.x); f[13] = f2bf(y1.y); f[14] = f2bf(y1.z); f[15] = f2bf(y1.w);
    return f;
}

// ---------------------------------------------------------------------------
// TDM: DMA one B K-chunk (256 emb rows x 32 fp32) into LDS via the Tensor
// Data Mover. 2-D descriptor: D# group0 (4 SGPRs) + group1 (8 SGPRs),
// VADDR2/3 = NULL. Tracked by TENSORcnt.
// ---------------------------------------------------------------------------
__device__ inline void tdm_load_b_chunk(const float* gsrc, unsigned ldsByteOff) {
    unsigned long long ga = (unsigned long long)(const void*)gsrc;  // byte address of tile start
    u32x4 g0;
    g0[0] = 1u;                                   // count=1 valid descriptor, user mode
    g0[1] = ldsByteOff;                           // lds_addr [63:32]
    g0[2] = (unsigned)ga;                         // global_addr[31:0]  -> bits [95:64]
    g0[3] = (unsigned)((ga >> 32) & 0x01FFFFFFu)  // global_addr[56:32] -> bits [120:96]
          | (2u << 30);                           // type=2 ("image")   -> bits [127:126]
    u32x8 g1;
    g1[0] = (2u << 16);                           // workgroup_mask=0, data_size=2 (4 bytes)
    g1[1] = ((unsigned)H_ & 0xFFFFu) << 16;       // tensor_dim0 lo16 at [63:48]
    g1[2] = ((unsigned)H_ >> 16)                  // tensor_dim0 hi16 at [79:64]
          | (((unsigned)V_ & 0xFFFFu) << 16);     // tensor_dim1 lo16 at [95:80]
    g1[3] = ((unsigned)V_ >> 16)                  // tensor_dim1 hi16 at [111:96]
          | (32u << 16);                          // tile_dim0 = 32 elements
    g1[4] = 256u;                                 // tile_dim1 = 256 rows, tile_dim2 = 0 (2D)
    g1[5] = (unsigned)H_;                         // tensor_dim0_stride lo32 = 4096 elements
    g1[6] = 0u;                                   // stride hi, dim1_stride (unused for 2D)
    g1[7] = 0u;
    asm volatile("tensor_load_to_lds %0, %1" :: "s"(g0), "s"(g1) : "memory");
}

// ---------------------------------------------------------------------------
// Kernel 1: logits = hidden @ emb^T + bias.  bf16 WMMA, fp32 accumulate.
// Block = 256 threads = 8 waves; covers C tile [mBase:+16, nBase0:+256].
// B panel K-chunks staged into LDS by TDM, double-buffered (2 x 32 KB of the
// 320 KB WGP LDS). Each wave: 1 shared A fragment -> 2 WMMAs.
// ---------------------------------------------------------------------------
__global__ __launch_bounds__(256) void logits_wmma_kernel(
    const float* __restrict__ hidden, const float* __restrict__ emb,
    const float* __restrict__ bias, float* __restrict__ logits)
{
    extern __shared__ float sB[];                 // 2 x (256 rows x 32 floats) = 64 KB
    const int lane = threadIdx.x & 31;
    const int wave = threadIdx.x >> 5;
    const int half = lane >> 4;
    const int r    = lane & 15;

    const int mBase  = blockIdx.x * 16;           // batch tile (fast dim -> B panel reuse in L2)
    const int nBase0 = blockIdx.y * 256;          // vocab panel

    const float* aRow  = hidden + (mBase + r) * H_;
    const float* bPane = emb + (size_t)nBase0 * H_;

    const int NCH = H_ / 32;                      // 128 K-chunks

    // Prologue: stage chunk 0 into buffer 0
    if (wave == 0) tdm_load_b_chunk(bPane, 0u);

    v8f acc0 = {};
    v8f acc1 = {};
    unsigned buf = 0u;                            // LDS byte offset of current buffer (0 / 32768)
    for (int kc = 0; kc < NCH; ++kc) {
        if (wave == 0) __builtin_amdgcn_s_wait_tensorcnt(0);  // chunk kc landed in LDS
        __syncthreads();
        if (wave == 0 && kc + 1 < NCH)                         // prefetch next chunk
            tdm_load_b_chunk(bPane + (kc + 1) * 32, buf ^ 32768u);

        const int k = kc * 32;
        v16bf af = make_frag(aRow + k, half);

        // B rows for this wave's two N-subtiles, fp32 in LDS (row-major, 32 floats/row)
        const float* b0 = sB + (buf >> 2) + (wave * 32 + r) * 32;
        const float* b1 = b0 + 16 * 32;
        v16bf bf0 = make_frag(b0, half);
        v16bf bf1 = make_frag(b1, half);

        acc0 = __builtin_amdgcn_wmma_f32_16x16x32_bf16(
            false, af, false, bf0, (short)0, acc0, false, false);
        acc1 = __builtin_amdgcn_wmma_f32_16x16x32_bf16(
            false, af, false, bf1, (short)0, acc1, false, false);

        __syncthreads();                          // all reads of this buffer done
        buf ^= 32768u;
    }

    // C layout: element v -> row M = v + half*8, col N = lane&15
    const int col0 = nBase0 + wave * 32 + r;
    const int col1 = col0 + 16;
    const float bv0 = bias[col0];
    const float bv1 = bias[col1];
#pragma unroll
    for (int v = 0; v < 8; ++v) {
        const int row = mBase + v + half * 8;
        logits[row * V_ + col0] = acc0[v] + bv0;
        logits[row * V_ + col1] = acc1[v] + bv1;
    }
}

// ---------------------------------------------------------------------------
// Kernel 2: presence/frequency penalties. One block per sequence, one thread
// per generated-token slot; O(L^2) duplicate counting in LDS (L=128).
// ---------------------------------------------------------------------------
__global__ __launch_bounds__(L_) void penalty_kernel(
    const int* __restrict__ tokens, const float* __restrict__ freq,
    const float* __restrict__ pres, float* __restrict__ logits)
{
    __shared__ int toks[L_];
    const int row = blockIdx.x;
    const int t   = threadIdx.x;
    toks[t] = tokens[row * L_ + t];
    __syncthreads();

    const int tok = toks[t];
    int cnt = 0;
    bool first = true;
    for (int j = 0; j < L_; ++j) {
        cnt += (toks[j] == tok);
        if (j < t && toks[j] == tok) first = false;   // only first occurrence applies
    }
    if (first)
        logits[row * V_ + tok] -= freq[row] * (float)cnt + pres[row];
}

// ---------------------------------------------------------------------------
// Kernel 3: temperature + top-k/top-p masking + softmax, one 1024-thread
// (32-wave) block per row. Full row of exp-values staged in 125 KB of LDS;
// thresholds found by bisection instead of a 32000-element sort.
// ---------------------------------------------------------------------------
template<bool MAX>
__device__ inline float block_reduce(float v, float* red, float* bcastp) {
    const int lane = threadIdx.x & 31;
    const int wid  = threadIdx.x >> 5;
#pragma unroll
    for (int off = 16; off > 0; off >>= 1) {
        float o = __shfl_down(v, off, 32);
        v = MAX ? fmaxf(v, o) : (v + o);
    }
    __syncthreads();                 // protect red[] reuse across calls
    if (lane == 0) red[wid] = v;
    __syncthreads();
    if (wid == 0) {
        float x = red[lane];
#pragma unroll
        for (int off = 16; off > 0; off >>= 1) {
            float o = __shfl_down(x, off, 32);
            x = MAX ? fmaxf(x, o) : (x + o);
        }
        if (lane == 0) *bcastp = x;
    }
    __syncthreads();
    return *bcastp;
}

__global__ __launch_bounds__(1024) void topk_topp_kernel(
    float* __restrict__ logits, const float* __restrict__ temps,
    const float* __restrict__ topps, const int* __restrict__ topks)
{
    extern __shared__ float s[];     // V_ floats: x/T, then exp(x/T - max)
    __shared__ float red[32];
    __shared__ float bcast;

    const int row  = blockIdx.x;
    const int tid  = threadIdx.x;
    const float invT = 1.0f / temps[row];
    const float p    = topps[row];
    const float k    = (float)topks[row];
    float* rowp = logits + row * V_;

    // Pass 1: scale by temperature into LDS, track max
    float lmax = -3.4e38f;
    for (int i = tid; i < V_; i += 1024) {
        float x = rowp[i] * invT;
        s[i] = x;
        lmax = fmaxf(lmax, x);
    }
    lmax = block_reduce<true>(lmax, red, &bcast);

    // Pass 2: e = exp(x - max) in LDS; total mass Z
    float lsum = 0.0f;
    for (int i = tid; i < V_; i += 1024) {
        float e = __expf(s[i] - lmax);
        s[i] = e;
        lsum += e;
    }
    const float Z = block_reduce<false>(lsum, red, &bcast);

    // Bisection 1 (top-k): threshold tk with count(e > tk) ~= k.  e in (0,1].
    float lo = 0.0f, hi = 1.0f;
    for (int it = 0; it < 28; ++it) {
        const float mid = 0.5f * (lo + hi);
        float c = 0.0f;
        for (int i = tid; i < V_; i += 1024) c += (s[i] > mid) ? 1.0f : 0.0f;
        c = block_reduce<false>(c, red, &bcast);
        if (c >= k) lo = mid; else hi = mid;
    }
    const float tk = lo;             // keep e > tk  (~k survivors)

    // Bisection 2 (top-p): smallest t with sum(e > t) <= p*Z
    lo = 0.0f; hi = 1.0f;
    const float pZ = p * Z;
    for (int it = 0; it < 28; ++it) {
        const float mid = 0.5f * (lo + hi);
        float g = 0.0f;
        for (int i = tid; i < V_; i += 1024) { float e = s[i]; g += (e > mid) ? e : 0.0f; }
        g = block_reduce<false>(g, red, &bcast);
        if (g > pZ) lo = mid; else hi = mid;
    }
    const float tp = hi;             // keep e >= tp  => mass above it <= p*Z

    // Pass 3: renormalize survivors, zero the rest
    float l2 = 0.0f;
    for (int i = tid; i < V_; i += 1024) {
        float e = s[i];
        bool keep = (e > tk) && (e >= tp);
        l2 += keep ? e : 0.0f;
    }
    const float Z2 = block_reduce<false>(l2, red, &bcast);
    const float rZ2 = (Z2 > 0.0f) ? (1.0f / Z2) : 0.0f;
    for (int i = tid; i < V_; i += 1024) {
        float e = s[i];
        bool keep = (e > tk) && (e >= tp);
        rowp[i] = keep ? e * rZ2 : 0.0f;
    }
}

// ---------------------------------------------------------------------------
extern "C" void kernel_launch(void* const* d_in, const int* in_sizes, int n_in,
                              void* d_out, int out_size, void* d_ws, size_t ws_size,
                              hipStream_t stream) {
    (void)in_sizes; (void)n_in; (void)out_size; (void)d_ws; (void)ws_size;
    const float* hidden = (const float*)d_in[0];
    const float* emb    = (const float*)d_in[1];
    const float* bias   = (const float*)d_in[2];
    const int*   toks   = (const int*)  d_in[3];
    const float* pres   = (const float*)d_in[4];
    const float* freq   = (const float*)d_in[5];
    const float* temps  = (const float*)d_in[6];
    const float* topps  = (const float*)d_in[7];
    const int*   topks  = (const int*)  d_in[8];
    float* out = (float*)d_out;          // used in-place: logits -> probs

    dim3 g1(B_ / 16, V_ / 256);          // 16 x 125 blocks; M fastest for B-panel L2 reuse
    logits_wmma_kernel<<<g1, 256, 2 * 32768, stream>>>(hidden, emb, bias, out);
    penalty_kernel<<<B_, L_, 0, stream>>>(toks, freq, pres, out);
    topk_topp_kernel<<<B_, 1024, V_ * sizeof(float), stream>>>(out, temps, topps, topks);
}
